// BitEncoder_77764677861393
// MI455X (gfx1250) — compile-verified, matrix operand order checked by reading, not run on hardware
//
#include <hip/hip_runtime.h>
#include <math.h>

// ---------------- model dims (from reference) ----------------
#define BATCH   8
#define DIMD    192
#define NHEADS  4
#define HDIM    48          // 192 / 4
#define SEQ     1001        // 320000 -> /5 /4 /4 /4
#define MROWS   (BATCH*SEQ) // 8008
#define FFND    384
#define OUTD    768
#define QTILES  63          // ceil(SEQ/16)
#define KCH     32          // key chunk for flash attention

// frontend lengths: L_out = floor((L+2p-k)/s)+1
#define L_IN  320000
#define L0O   64001
#define L1O   16001
#define L2O   4001
#define L3O   1001

typedef _Float16 h16;
typedef __attribute__((ext_vector_type(16))) _Float16 v16h;
typedef __attribute__((ext_vector_type(8)))  _Float16 v8h;
typedef __attribute__((ext_vector_type(8)))  float    v8f;
typedef __attribute__((ext_vector_type(4)))  float    v4f;

__device__ __forceinline__ float gelu_f(float x) {
  float x3 = x * x * x;
  return 0.5f * x * (1.f + tanhf(0.7978845608028654f * (x + 0.044715f * x3)));
}

__device__ __forceinline__ v8f wmma_f16(v16h a, v16h b, v8f c) {
  return __builtin_amdgcn_wmma_f32_16x16x32_f16(false, a, false, b, (short)0, c, false, false);
}

// load a 16-element f16 fragment: 8 halfs at p, 8 halfs at p+16 (ISA 7.12.2 K split)
__device__ __forceinline__ v16h ldfrag_h(const h16* p) {
  v8h x0 = *(const v8h*)p;
  v8h x1 = *(const v8h*)(p + 16);
  v16h f;
#pragma unroll
  for (int i = 0; i < 8; ++i) { f[i] = x0[i]; f[8 + i] = x1[i]; }
  return f;
}

__device__ __forceinline__ v16h ldfrag_f(const float* p) {
  v4f a0 = *(const v4f*)p, a1 = *(const v4f*)(p + 4);
  v4f a2 = *(const v4f*)(p + 16), a3 = *(const v4f*)(p + 20);
  v16h f;
#pragma unroll
  for (int i = 0; i < 4; ++i) {
    f[i] = (h16)a0[i]; f[4 + i] = (h16)a1[i];
    f[8 + i] = (h16)a2[i]; f[12 + i] = (h16)a3[i];
  }
  return f;
}

// ---------------- ternary quantization (one block per tensor) ----------------
__global__ __launch_bounds__(256) void ternary_f16_kernel(const float* __restrict__ w,
                                                          h16* __restrict__ q, int n) {
  __shared__ float red[256];
  float s = 0.f;
  for (int i = threadIdx.x; i < n; i += 256) s += fabsf(w[i]);
  red[threadIdx.x] = s; __syncthreads();
  for (int o = 128; o; o >>= 1) {
    if ((int)threadIdx.x < o) red[threadIdx.x] += red[threadIdx.x + o];
    __syncthreads();
  }
  float scale = fmaxf(red[0] / (float)n, 1e-5f);
  for (int i = threadIdx.x; i < n; i += 256) {
    float t = rintf(w[i] / scale);
    t = fminf(fmaxf(t, -1.f), 1.f);
    q[i] = (h16)(t * scale);
  }
}

__global__ __launch_bounds__(256) void ternary_f32_kernel(const float* __restrict__ w,
                                                          float* __restrict__ q, int n) {
  __shared__ float red[256];
  float s = 0.f;
  for (int i = threadIdx.x; i < n; i += 256) s += fabsf(w[i]);
  red[threadIdx.x] = s; __syncthreads();
  for (int o = 128; o; o >>= 1) {
    if ((int)threadIdx.x < o) red[threadIdx.x] += red[threadIdx.x + o];
    __syncthreads();
  }
  float scale = fmaxf(red[0] / (float)n, 1e-5f);
  for (int i = threadIdx.x; i < n; i += 256) {
    float t = rintf(w[i] / scale);
    t = fminf(fmaxf(t, -1.f), 1.f);
    q[i] = t * scale;
  }
}

__global__ __launch_bounds__(256) void zerofill_kernel(float* __restrict__ p, int n) {
  for (int i = threadIdx.x; i < n; i += 256) p[i] = 0.f;
}

// ---------------- frontend conv0 (ci=1,k=10,s=5) -> (B,T,C) layout -----------
__global__ __launch_bounds__(256) void conv0_kernel(const float* __restrict__ x,
    const float* __restrict__ w, float* __restrict__ out) {
  int idx = blockIdx.x * 256 + threadIdx.x;
  int total = BATCH * L0O * 32;
  if (idx >= total) return;
  int c = idx % 32;
  int t = (idx / 32) % L0O;
  int b = idx / (32 * L0O);
  int base = t * 5 - 5;
  float acc = 0.f;
#pragma unroll
  for (int k = 0; k < 10; ++k) {
    int xx = base + k;
    if (xx >= 0 && xx < L_IN) acc += w[c * 10 + k] * x[(size_t)b * L_IN + xx];
  }
  out[idx] = acc;
}

// ---------------- im2col for k=8,s=4,p=4 convs: (B,T,C) f32 -> [B*TOUT, CI*8] f16
__global__ __launch_bounds__(256) void im2col_kernel(const float* __restrict__ in,
    h16* __restrict__ out, int CI, int TIN, int TOUT) {
  size_t idx = (size_t)blockIdx.x * 256 + threadIdx.x;
  size_t total = (size_t)BATCH * TOUT * CI * 8;
  if (idx >= total) return;
  int k  = (int)(idx & 7);
  int ci = (int)((idx >> 3) % (size_t)CI);
  size_t m = idx / ((size_t)CI * 8);
  int t = (int)(m % (size_t)TOUT);
  int b = (int)(m / (size_t)TOUT);
  int x = t * 4 + k - 4;
  float v = (x >= 0 && x < TIN) ? in[((size_t)b * TIN + x) * CI + ci] : 0.f;
  out[idx] = (h16)v;
}

// ---------------- fused GroupNorm(1 group over all C,T) + snake, (B,T,C) -----
__global__ __launch_bounds__(256) void gn_snake_kernel(float* __restrict__ buf,
    const float* __restrict__ g, const float* __restrict__ bb,
    const float* __restrict__ sa, const float* __restrict__ sp, int C, int T) {
  __shared__ float r1[256], r2[256];
  size_t N = (size_t)C * T;
  float* p = buf + (size_t)blockIdx.x * N;
  float s = 0.f, s2 = 0.f;
  for (size_t i = threadIdx.x; i < N; i += 256) { float v = p[i]; s += v; s2 += v * v; }
  r1[threadIdx.x] = s; r2[threadIdx.x] = s2; __syncthreads();
  for (int o = 128; o; o >>= 1) {
    if ((int)threadIdx.x < o) { r1[threadIdx.x] += r1[threadIdx.x + o]; r2[threadIdx.x] += r2[threadIdx.x + o]; }
    __syncthreads();
  }
  float mean = r1[0] / (float)N;
  float var  = r2[0] / (float)N - mean * mean;
  float inv  = rsqrtf(var + 1e-5f);
  for (size_t i = threadIdx.x; i < N; i += 256) {
    int c = (int)(i % (size_t)C);
    float v = (p[i] - mean) * inv * g[c] + bb[c];
    float a = sa[c];
    float sn = sinf(a * v + sp[c]);
    p[i] = v + sn * sn / (a + 1e-9f);
  }
}

// ---------------- sinusoidal positional embedding add ----------------
__global__ __launch_bounds__(256) void posemb_kernel(float* __restrict__ x) {
  int idx = blockIdx.x * 256 + threadIdx.x;
  int total = MROWS * DIMD;
  if (idx >= total) return;
  int d = idx % DIMD;
  int s = (idx / DIMD) % SEQ;
  int i2 = d & ~1;
  float div = expf(-(float)i2 * (9.210340371976184f / (float)DIMD)); // ln(10000)
  float arg = (float)s * div;
  x[idx] += (d & 1) ? cosf(arg) : sinf(arg);
}

// ---------------- LayerNorm: one wave per row; OH=1 -> f16 output ----------------
template<int OH>
__global__ __launch_bounds__(256) void layernorm_kernel(const float* __restrict__ in,
    void* __restrict__ outv, const float* __restrict__ g, const float* __restrict__ b,
    int M, int D) {
  int row = blockIdx.x * 8 + (threadIdx.x >> 5);
  if (row >= M) return;
  int lane = threadIdx.x & 31;
  const float* x = in + (size_t)row * D;
  float s = 0.f, s2 = 0.f;
  for (int d = lane; d < D; d += 32) { float v = x[d]; s += v; s2 += v * v; }
  for (int m = 16; m; m >>= 1) { s += __shfl_xor(s, m); s2 += __shfl_xor(s2, m); }
  float mean = s / (float)D;
  float inv  = rsqrtf(s2 / (float)D - mean * mean + 1e-5f);
  if (OH) {
    h16* o = (h16*)outv + (size_t)row * D;
    for (int d = lane; d < D; d += 32) o[d] = (h16)((x[d] - mean) * inv * g[d] + b[d]);
  } else {
    float* o = (float*)outv + (size_t)row * D;
    for (int d = lane; d < D; d += 32) o[d] = (x[d] - mean) * inv * g[d] + b[d];
  }
}

// ---------------- WMMA GEMM: out[M,N] = A[M,K] @ W[N,K]^T (+bias, fusions) ----
// One wave computes a 16x64 block (4 accumulators). AH: A f16; OH: out f16.
// FUSE bit0: gelu ; bit1: out = resid + alpha*(acc+bias)
template<int FUSE, int AH, int OH>
__global__ __launch_bounds__(256) void gemm_wmma_kernel(
    const void* __restrict__ Av, const h16* __restrict__ W,
    const float* __restrict__ bias, const float* __restrict__ resid,
    float alpha, void* __restrict__ outv, int M, int N, int K) {
  const int wave = threadIdx.x >> 5;
  const int lane = threadIdx.x & 31;
  const int tilesN = N >> 6;                  // 64-wide column blocks
  const int tiles  = ((M + 15) >> 4) * tilesN;
  const int tile   = blockIdx.x * 8 + wave;
  if (tile >= tiles) return;
  const int m0 = (tile / tilesN) << 4;
  const int n0 = (tile % tilesN) << 6;
  const int lg = lane >> 4;
  const int lr = lane & 15;

  int ma = m0 + lr; if (ma > M - 1) ma = M - 1;   // clamp loads; stores guarded

  const size_t wstride = (size_t)16 * K;
  v8f acc0 = {}, acc1 = {}, acc2 = {}, acc3 = {};
  for (int k0 = 0; k0 < K; k0 += 32) {
    v16h af;
    if (AH) af = ldfrag_h((const h16*)Av + (size_t)ma * K + k0 + (lg << 3));
    else    af = ldfrag_f((const float*)Av + (size_t)ma * K + k0 + (lg << 3));
    const h16* wp = W + (size_t)(n0 + lr) * K + k0 + (lg << 3);
    __builtin_prefetch(wp + 32, 0, 1);   // global_prefetch_b8 on weight stream
    v16h b0 = ldfrag_h(wp);
    v16h b1 = ldfrag_h(wp + wstride);
    v16h b2 = ldfrag_h(wp + 2 * wstride);
    v16h b3 = ldfrag_h(wp + 3 * wstride);
    acc0 = wmma_f16(af, b0, acc0);
    acc1 = wmma_f16(af, b1, acc1);
    acc2 = wmma_f16(af, b2, acc2);
    acc3 = wmma_f16(af, b3, acc3);
  }

#pragma unroll
  for (int r = 0; r < 8; ++r) {
    const int m = m0 + r + (lg << 3);
    if (m < M) {
      const size_t base = (size_t)m * N;
#pragma unroll
      for (int c = 0; c < 4; ++c) {
        const int n = n0 + (c << 4) + lr;
        float v = (c == 0 ? acc0[r] : c == 1 ? acc1[r] : c == 2 ? acc2[r] : acc3[r]) + bias[n];
        if (FUSE & 1) v = gelu_f(v);
        if (FUSE & 2) v = resid[base + n] + alpha * v;
        if (OH) ((h16*)outv)[base + n] = (h16)v;
        else    ((float*)outv)[base + n] = v;
      }
    }
  }
}

// ---------------- elementwise snake, (rows, C) channel-fast ----------------
__global__ __launch_bounds__(256) void snake_kernel(float* __restrict__ buf,
    const float* __restrict__ a, const float* __restrict__ p, int C, int total) {
  int i = blockIdx.x * 256 + threadIdx.x;
  if (i >= total) return;
  int c = i % C;
  float v = buf[i];
  float sn = sinf(a[c] * v + p[c]);
  buf[i] = v + sn * sn / (a[c] + 1e-9f);
}

// ---------------- depthwise conv k=15 along S, (B,S,C) layout ----------------
__global__ __launch_bounds__(256) void dwconv_kernel(const float* __restrict__ in,
    const float* __restrict__ w, const float* __restrict__ bias,
    float* __restrict__ out) {
  int idx = blockIdx.x * 256 + threadIdx.x;
  int total = MROWS * DIMD;
  if (idx >= total) return;
  int c = idx % DIMD;
  int s = (idx / DIMD) % SEQ;
  int b0 = idx / (DIMD * SEQ);
  float acc = bias[c];
  const float* wp = w + c * 15;
#pragma unroll
  for (int k = 0; k < 15; ++k) {
    int ss = s + k - 7;
    if (ss >= 0 && ss < SEQ)
      acc += wp[k] * in[((size_t)b0 * SEQ + ss) * DIMD + c];
  }
  out[idx] = acc;
}

// ---------------- WMMA flash attention -------------------------------------
__global__ __launch_bounds__(256) void attention_wmma_kernel(const h16* __restrict__ qkv,
                                                             h16* __restrict__ out) {
  __shared__ __align__(16) h16 Pl[8][16 * KCH];   // 1 KB / wave
  __shared__ __align__(16) h16 Vt[8][HDIM * KCH]; // 3 KB / wave
  const int wave = threadIdx.x >> 5;
  const int lane = threadIdx.x & 31;
  const int lg = lane >> 4, lr = lane & 15;
  const int rid = blockIdx.x * 8 + wave;          // exactly B*H*QTILES waves
  const int b0 = rid / (NHEADS * QTILES);
  const int rem = rid % (NHEADS * QTILES);
  const int h = rem / QTILES;
  const int qt = rem % QTILES;
  const int q0 = qt * 16;

  h16* Pw = Pl[wave];
  h16* Vw = Vt[wave];
  const size_t rowstride = 576;
  const size_t bbase = (size_t)b0 * SEQ * rowstride;

  int qr = q0 + lr; if (qr > SEQ - 1) qr = SEQ - 1;
  const h16* qp = qkv + bbase + (size_t)qr * rowstride + h * HDIM;
  v16h qf0, qf1;
  {
    v8h a = *(const v8h*)(qp + (lg << 3));
    v8h b = *(const v8h*)(qp + 16 + (lg << 3));
    v8h c = *(const v8h*)(qp + 32 + (lg << 3));
#pragma unroll
    for (int i = 0; i < 8; ++i) { qf0[i] = a[i]; qf0[8 + i] = b[i]; qf1[i] = c[i]; qf1[8 + i] = (h16)0.f; }
  }

  const float scale = 0.14433756729740643f;  // 1/sqrt(48)
  float m[8], l[8];
  v8f o0 = {}, o1 = {}, o2 = {};
#pragma unroll
  for (int r = 0; r < 8; ++r) { m[r] = -3.0e38f; l[r] = 0.f; }

  for (int c0 = 0; c0 < SEQ; c0 += KCH) {
    v8f s0 = {}, s1 = {};
    {
      int kr = c0 + lr; if (kr > SEQ - 1) kr = SEQ - 1;
      const h16* kp = qkv + bbase + (size_t)kr * rowstride + DIMD + h * HDIM;
      v8h a = *(const v8h*)(kp + (lg << 3));
      v8h b = *(const v8h*)(kp + 16 + (lg << 3));
      v8h c = *(const v8h*)(kp + 32 + (lg << 3));
      v16h kf0, kf1;
#pragma unroll
      for (int i = 0; i < 8; ++i) { kf0[i] = a[i]; kf0[8 + i] = b[i]; kf1[i] = c[i]; kf1[8 + i] = (h16)0.f; }
      s0 = wmma_f16(qf0, kf0, s0);
      s0 = wmma_f16(qf1, kf1, s0);
    }
    {
      int kr = c0 + 16 + lr; if (kr > SEQ - 1) kr = SEQ - 1;
      const h16* kp = qkv + bbase + (size_t)kr * rowstride + DIMD + h * HDIM;
      v8h a = *(const v8h*)(kp + (lg << 3));
      v8h b = *(const v8h*)(kp + 16 + (lg << 3));
      v8h c = *(const v8h*)(kp + 32 + (lg << 3));
      v16h kf0, kf1;
#pragma unroll
      for (int i = 0; i < 8; ++i) { kf0[i] = a[i]; kf0[8 + i] = b[i]; kf1[i] = c[i]; kf1[8 + i] = (h16)0.f; }
      s1 = wmma_f16(qf0, kf0, s1);
      s1 = wmma_f16(qf1, kf1, s1);
    }
    {
      int vr = c0 + lane; if (vr > SEQ - 1) vr = SEQ - 1;
      const h16* vp = qkv + bbase + (size_t)vr * rowstride + 2 * DIMD + h * HDIM;
#pragma unroll
      for (int d = 0; d < HDIM; d += 8) {
        v8h v = *(const v8h*)(vp + d);
#pragma unroll
        for (int i = 0; i < 8; ++i) Vw[(d + i) * KCH + lane] = v[i];
      }
    }
    const int key0 = c0 + lr, key1 = c0 + 16 + lr;
#pragma unroll
    for (int r = 0; r < 8; ++r) {
      float a0 = (key0 < SEQ) ? s0[r] : -3.0e38f;
      float a1 = (key1 < SEQ) ? s1[r] : -3.0e38f;
      float rm = fmaxf(a0, a1);
      rm = fmaxf(rm, __shfl_xor(rm, 1));
      rm = fmaxf(rm, __shfl_xor(rm, 2));
      rm = fmaxf(rm, __shfl_xor(rm, 4));
      rm = fmaxf(rm, __shfl_xor(rm, 8));
      float mn = fmaxf(m[r], rm);
      float fac = __expf(scale * (m[r] - mn));
      float p0 = __expf(scale * (a0 - mn));
      float p1 = __expf(scale * (a1 - mn));
      float rs = p0 + p1;
      rs += __shfl_xor(rs, 1);
      rs += __shfl_xor(rs, 2);
      rs += __shfl_xor(rs, 4);
      rs += __shfl_xor(rs, 8);
      l[r] = l[r] * fac + rs;
      m[r] = mn;
      o0[r] *= fac; o1[r] *= fac; o2[r] *= fac;
      const int row = (lg << 3) + r;
      Pw[row * KCH + lr] = (h16)p0;
      Pw[row * KCH + 16 + lr] = (h16)p1;
    }
    v16h pf  = ldfrag_h(Pw + lr * KCH + (lg << 3));
    v16h vb0 = ldfrag_h(Vw + (0 * 16 + lr) * KCH + (lg << 3));
    v16h vb1 = ldfrag_h(Vw + (1 * 16 + lr) * KCH + (lg << 3));
    v16h vb2 = ldfrag_h(Vw + (2 * 16 + lr) * KCH + (lg << 3));
    o0 = wmma_f16(pf, vb0, o0);
    o1 = wmma_f16(pf, vb1, o1);
    o2 = wmma_f16(pf, vb2, o2);
  }

#pragma unroll
  for (int r = 0; r < 8; ++r) {
    const int row = q0 + (lg << 3) + r;
    if (row < SEQ) {
      const float inv = 1.f / l[r];
      const size_t ob = ((size_t)b0 * SEQ + row) * DIMD + h * HDIM + lr;
      out[ob]      = (h16)(o0[r] * inv);
      out[ob + 16] = (h16)(o1[r] * inv);
      out[ob + 32] = (h16)(o2[r] * inv);
    }
  }
}

// ======================= host side =======================

enum { B_FF1_LN_G = 0, B_FF1_LN_B, B_FF1_W1, B_FF1_B1, B_FF1_W2, B_FF1_B2,
       B_ATTN_LN_G, B_ATTN_LN_B, B_WQKV, B_BQKV, B_WO, B_BO,
       B_CONV_LN_G, B_CONV_LN_B, B_PW1_W, B_PW1_B, B_SN1_A, B_SN1_P,
       B_DW_W, B_DW_B, B_GN_G, B_GN_B, B_SN2_A, B_SN2_P, B_PW2_W, B_PW2_B,
       B_FF2_LN_G, B_FF2_LN_B, B_FF2_W1, B_FF2_B1, B_FF2_W2, B_FF2_B2,
       B_FINAL_LN_G, B_FINAL_LN_B };

extern "C" void kernel_launch(void* const* d_in, const int* in_sizes, int n_in,
                              void* d_out, int out_size, void* d_ws, size_t ws_size,
                              hipStream_t stream) {
  (void)in_sizes; (void)n_in; (void)out_size; (void)ws_size;
  auto F = [&](int i) -> const float* { return (const float*)d_in[i]; };
  const int FE = 1, BK = 21, GIDX = 157;

  char* ws = (char*)d_ws;
  size_t off = 0;
  auto alloc = [&](size_t nbytes) -> void* {
    off = (off + 255) & ~(size_t)255;
    void* p = ws + off;
    off += nbytes;
    return p;
  };

  // -------- workspace --------
  // FAb: F0 (B,T,C 8x64001x32) -> later F2 (8x4001x128) -> later head hidden HB
  float* FAb = (float*)alloc(sizeof(float) * (size_t)BATCH * L0O * 32);
  // FBb: F1 (8x16001x64) -> later QKV/ATT (f16)
  float* FBb = (float*)alloc(sizeof(float) * (size_t)BATCH * L1O * 64);
  // arena: im2col staging (frontend phase), then transformer temporaries
  char* AR = (char*)alloc(sizeof(h16) * (size_t)BATCH * L1O * 256);  // 65.5 MB
  float* X0f = (float*)alloc(sizeof(float) * (size_t)MROWS * 256);   // frontend out

  h16*   Abuf = (h16*)AR;                      // im2col buffer (frontend only)
  float* XA = (float*)AR;                      // transformer phase aliases
  float* XB = XA + (size_t)MROWS * DIMD;
  float* T2 = XB + (size_t)MROWS * DIMD;
  float* T3 = T2 + (size_t)MROWS * FFND;
  h16*  T1h = (h16*)(T3 + (size_t)MROWS * DIMD);
  h16*  T2h = T1h + (size_t)MROWS * DIMD;

  float* HB  = FAb;                            // [MROWS,768] f32 head hidden
  h16*  QKV  = (h16*)FBb;                      // [MROWS,576] f16
  h16*  ATTh = QKV + (size_t)MROWS * 3 * DIMD; // [MROWS,192] f16

  // quantized weights
  float* FW0q = (float*)alloc(sizeof(float) * 32 * 10);
  h16* FW1q = (h16*)alloc(sizeof(h16) * 64 * 256);
  h16* FW2q = (h16*)alloc(sizeof(h16) * 128 * 512);
  h16* FW3q = (h16*)alloc(sizeof(h16) * 256 * 1024);
  float* DWQ[4];
  for (int j = 0; j < 4; ++j) DWQ[j] = (float*)alloc(sizeof(float) * DIMD * 15);
  h16* inWq  = (h16*)alloc(sizeof(h16) * DIMD * 256);
  h16* opW1q = (h16*)alloc(sizeof(h16) * FFND * DIMD);
  h16* opW2q = (h16*)alloc(sizeof(h16) * OUTD * FFND);
  h16 *ff1w1q[4], *ff1w2q[4], *wqkvq[4], *woq[4], *pw1q[4], *pw2q[4], *ff2w1q[4], *ff2w2q[4];
  for (int j = 0; j < 4; ++j) {
    ff1w1q[j] = (h16*)alloc(sizeof(h16) * FFND * DIMD);
    ff1w2q[j] = (h16*)alloc(sizeof(h16) * DIMD * FFND);
    wqkvq[j]  = (h16*)alloc(sizeof(h16) * 3 * DIMD * DIMD);
    woq[j]    = (h16*)alloc(sizeof(h16) * DIMD * DIMD);
    pw1q[j]   = (h16*)alloc(sizeof(h16) * DIMD * DIMD);
    pw2q[j]   = (h16*)alloc(sizeof(h16) * DIMD * DIMD);
    ff2w1q[j] = (h16*)alloc(sizeof(h16) * FFND * DIMD);
    ff2w2q[j] = (h16*)alloc(sizeof(h16) * DIMD * FFND);
  }
  float* zbias = (float*)alloc(sizeof(float) * 256);

  auto quantH = [&](const float* w, h16* q, int n) {
    ternary_f16_kernel<<<1, 256, 0, stream>>>(w, q, n);
  };
  auto quantF = [&](const float* w, float* q, int n) {
    ternary_f32_kernel<<<1, 256, 0, stream>>>(w, q, n);
  };
  auto gemm = [&](int fuse, int ah, int oh, const void* A, const h16* W,
                  const float* bias, const float* resid, float alpha, void* out,
                  int M, int N, int K) {
    int tiles = ((M + 15) / 16) * (N / 64);
    int blocks = (tiles + 7) / 8;
#define GO(Fz, Az, Oz) gemm_wmma_kernel<Fz, Az, Oz><<<blocks, 256, 0, stream>>>(A, W, bias, resid, alpha, out, M, N, K)
    if      (fuse == 0 && ah == 0 && oh == 0) GO(0, 0, 0);
    else if (fuse == 0 && ah == 1 && oh == 0) GO(0, 1, 0);
    else if (fuse == 0 && ah == 1 && oh == 1) GO(0, 1, 1);
    else if (fuse == 1 && ah == 1 && oh == 1) GO(1, 1, 1);
    else if (fuse == 1 && ah == 0 && oh == 1) GO(1, 0, 1);
    else if (fuse == 2 && ah == 1 && oh == 0) GO(2, 1, 0);
    else                                      GO(2, 0, 0);
#undef GO
  };
  auto lnH = [&](const float* in, h16* out, const float* g, const float* b, int M, int D) {
    layernorm_kernel<1><<<(M + 7) / 8, 256, 0, stream>>>(in, out, g, b, M, D);
  };
  auto lnF = [&](const float* in, float* out, const float* g, const float* b, int M, int D) {
    layernorm_kernel<0><<<(M + 7) / 8, 256, 0, stream>>>(in, out, g, b, M, D);
  };
  auto grid = [](int total) { return (total + 255) / 256; };
  auto gridl = [](size_t total) { return (int)((total + 255) / 256); };

  // -------- quantize all ternary weights; zero bias --------
  zerofill_kernel<<<1, 256, 0, stream>>>(zbias, 256);
  quantF(F(FE + 0),  FW0q, 32 * 10);
  quantH(F(FE + 5),  FW1q, 64 * 256);
  quantH(F(FE + 10), FW2q, 128 * 512);
  quantH(F(FE + 15), FW3q, 256 * 1024);
  for (int j = 0; j < 4; ++j) {
    const int bb = BK + 34 * j;
    quantH(F(bb + B_FF1_W1), ff1w1q[j], FFND * DIMD);
    quantH(F(bb + B_FF1_W2), ff1w2q[j], DIMD * FFND);
    quantH(F(bb + B_WQKV),  wqkvq[j],  3 * DIMD * DIMD);
    quantH(F(bb + B_WO),    woq[j],    DIMD * DIMD);
    quantH(F(bb + B_PW1_W), pw1q[j],   DIMD * DIMD);
    quantH(F(bb + B_PW2_W), pw2q[j],   DIMD * DIMD);
    quantH(F(bb + B_FF2_W1), ff2w1q[j], FFND * DIMD);
    quantH(F(bb + B_FF2_W2), ff2w2q[j], DIMD * FFND);
    quantF(F(bb + B_DW_W), DWQ[j], DIMD * 15);
  }
  quantH(F(GIDX + 0), inWq,  DIMD * 256);
  quantH(F(GIDX + 2), opW1q, FFND * DIMD);
  quantH(F(GIDX + 4), opW2q, OUTD * FFND);

  // -------- frontend: conv0 direct, convs 1-3 as im2col + WMMA GEMM --------
  const float* x = F(0);
  float* F0 = FAb;   // (B, L0O, 32)
  float* F1 = FBb;   // (B, L1O, 64)
  float* F2 = FAb;   // (B, L2O, 128)  (F0 dead after its im2col)
  conv0_kernel<<<grid(BATCH * L0O * 32), 256, 0, stream>>>(x, FW0q, F0);
  gn_snake_kernel<<<BATCH, 256, 0, stream>>>(F0, F(FE + 1), F(FE + 2), F(FE + 3), F(FE + 4), 32, L0O);
  im2col_kernel<<<gridl((size_t)BATCH * L1O * 256), 256, 0, stream>>>(F0, Abuf, 32, L0O, L1O);
  gemm(0, 1, 0, Abuf, FW1q, zbias, nullptr, 0.f, F1, BATCH * L1O, 64, 256);
  gn_snake_kernel<<<BATCH, 256, 0, stream>>>(F1, F(FE + 6), F(FE + 7), F(FE + 8), F(FE + 9), 64, L1O);
  im2col_kernel<<<gridl((size_t)BATCH * L2O * 512), 256, 0, stream>>>(F1, Abuf, 64, L1O, L2O);
  gemm(0, 1, 0, Abuf, FW2q, zbias, nullptr, 0.f, F2, BATCH * L2O, 128, 512);
  gn_snake_kernel<<<BATCH, 256, 0, stream>>>(F2, F(FE + 11), F(FE + 12), F(FE + 13), F(FE + 14), 128, L2O);
  im2col_kernel<<<gridl((size_t)BATCH * L3O * 1024), 256, 0, stream>>>(F2, Abuf, 128, L2O, L3O);
  gemm(0, 1, 0, Abuf, FW3q, zbias, nullptr, 0.f, X0f, BATCH * L3O, 256, 1024);
  gn_snake_kernel<<<BATCH, 256, 0, stream>>>(X0f, F(FE + 16), F(FE + 17), F(FE + 18), F(FE + 19), 256, L3O);

  // -------- input projection + positional embedding --------
  gemm(0, 0, 0, X0f, inWq, F(GIDX + 1), nullptr, 0.f, XA, MROWS, DIMD, 256);
  posemb_kernel<<<grid(MROWS * DIMD), 256, 0, stream>>>(XA);

  // -------- transformer blocks --------
  float* cur = XA;
  float* nxt = XB;
  for (int j = 0; j < 4; ++j) {
    const int bb = BK + 34 * j;
    lnH(cur, T1h, F(bb + B_FF1_LN_G), F(bb + B_FF1_LN_B), MROWS, DIMD);
    gemm(1, 1, 1, T1h, ff1w1q[j], F(bb + B_FF1_B1), nullptr, 0.f, T2h, MROWS, FFND, DIMD);
    gemm(2, 1, 0, T2h, ff1w2q[j], F(bb + B_FF1_B2), cur, 0.5f, nxt, MROWS, DIMD, FFND);
    { float* t = cur; cur = nxt; nxt = t; }
    lnH(cur, T1h, F(bb + B_ATTN_LN_G), F(bb + B_ATTN_LN_B), MROWS, DIMD);
    gemm(0, 1, 1, T1h, wqkvq[j], F(bb + B_BQKV), nullptr, 0.f, QKV, MROWS, 3 * DIMD, DIMD);
    attention_wmma_kernel<<<(BATCH * NHEADS * QTILES) / 8, 256, 0, stream>>>(QKV, ATTh);
    gemm(2, 1, 0, ATTh, woq[j], F(bb + B_BO), cur, 1.f, nxt, MROWS, DIMD, DIMD);
    { float* t = cur; cur = nxt; nxt = t; }
    lnH(cur, T1h, F(bb + B_CONV_LN_G), F(bb + B_CONV_LN_B), MROWS, DIMD);
    gemm(0, 1, 0, T1h, pw1q[j], F(bb + B_PW1_B), nullptr, 0.f, T2, MROWS, DIMD, DIMD);
    snake_kernel<<<grid(MROWS * DIMD), 256, 0, stream>>>(T2, F(bb + B_SN1_A), F(bb + B_SN1_P), DIMD, MROWS * DIMD);
    dwconv_kernel<<<grid(MROWS * DIMD), 256, 0, stream>>>(T2, DWQ[j], F(bb + B_DW_B), T3);
    gn_snake_kernel<<<BATCH, 256, 0, stream>>>(T3, F(bb + B_GN_G), F(bb + B_GN_B),
                                               F(bb + B_SN2_A), F(bb + B_SN2_P), DIMD, SEQ);
    gemm(2, 0, 0, T3, pw2q[j], F(bb + B_PW2_B), cur, 1.f, nxt, MROWS, DIMD, DIMD);
    { float* t = cur; cur = nxt; nxt = t; }
    lnH(cur, T1h, F(bb + B_FF2_LN_G), F(bb + B_FF2_LN_B), MROWS, DIMD);
    gemm(1, 1, 1, T1h, ff2w1q[j], F(bb + B_FF2_B1), nullptr, 0.f, T2h, MROWS, FFND, DIMD);
    gemm(2, 1, 0, T2h, ff2w2q[j], F(bb + B_FF2_B2), cur, 0.5f, nxt, MROWS, DIMD, FFND);
    { float* t = cur; cur = nxt; nxt = t; }
    lnF(cur, nxt, F(bb + B_FINAL_LN_G), F(bb + B_FINAL_LN_B), MROWS, DIMD);
    { float* t = cur; cur = nxt; nxt = t; }
  }

  // -------- output head --------
  gemm(1, 0, 1, cur, opW1q, F(GIDX + 3), nullptr, 0.f, T2h, MROWS, FFND, DIMD);
  gemm(0, 1, 0, T2h, opW2q, F(GIDX + 5), nullptr, 0.f, HB, MROWS, OUTD, FFND);
  lnF(HB, (float*)d_out, F(GIDX + 6), F(GIDX + 7), MROWS, OUTD);
}
